// RNNPytorch_59820304498848
// MI455X (gfx1250) — matrix operand order repaired and checked by reading, not compiled, not run on hardware
//
#include <hip/hip_runtime.h>
#include <hip/hip_bf16.h>

// ---------------------------------------------------------------------------
// RNN (leaky-ReLU) for MI455X / gfx1250, wave32 + WMMA f16->f32.
//   B=64, T=512, D_IN=D_H=D_OUT=1024
// Phases:
//   1) cvt x->f16 ; transpose+cvt W_xh/W_hh/W_ho to [N x K] f16 ; cvt h->f16
//   2) GEMM: xW = x16 @ W_xh + b_h   (f32, stashed in outputs half of d_out)
//   3) persistent scan kernel (8 blocks, global barrier per step):
//         h = leaky_relu(xW_t + h @ W_hh); write hiddens(f32) + hiddens16
//   4) GEMM: outputs = hiddens16 @ W_ho + b_o  (overwrites xW stash)
// ---------------------------------------------------------------------------

#define RNN_B   64
#define RNN_T   512
#define RNN_D   1024   // D_IN == D_H == D_OUT

typedef __attribute__((ext_vector_type(16))) _Float16 v16h;
typedef __attribute__((ext_vector_type(8)))  _Float16 v8h;
typedef __attribute__((ext_vector_type(8)))  float    v8f;

// ---- WMMA fragment loaders (layouts per CDNA5 ISA §7.12.2) -----

// A: 16x32 f16 tile, row-major source. lanes 0-15: M=lane, K in
// {8g..8g+7} -> elems 0..7, {16+8g..16+8g+7} -> elems 8..15  (g = lane>=16)
__device__ __forceinline__ v16h load_a_frag(const _Float16* __restrict__ A,
                                            int lda, int m0, int k0) {
  int lane = threadIdx.x & 31;
  int g    = lane >> 4;
  int r    = lane & 15;
  const _Float16* p = A + (size_t)(m0 + r) * lda + k0 + 8 * g;
  v8h lo = *(const v8h*)p;          // K = k0+8g   .. +7
  v8h hi = *(const v8h*)(p + 16);   // K = k0+16+8g.. +7
  return __builtin_shufflevector(lo, hi, 0,1,2,3,4,5,6,7,8,9,10,11,12,13,14,15);
}

// B: 32x16 f16 tile. Source is the PRE-TRANSPOSED weight Wt[n*ldk + k], so
// each lane (= output column n) reads two runs of 8 contiguous K values:
// lanes 0-15: N=lane, K = k0+8g + (0..7), k0+16+8g + (0..7)  (g = lane>=16)
__device__ __forceinline__ v16h load_b_frag(const _Float16* __restrict__ Wt,
                                            int ldk, int n0, int k0) {
  int lane = threadIdx.x & 31;
  int g    = lane >> 4;
  int n    = lane & 15;
  const _Float16* p = Wt + (size_t)(n0 + n) * ldk + k0 + 8 * g;
  v8h lo = *(const v8h*)p;          // K = k0+8g   .. +7
  v8h hi = *(const v8h*)(p + 16);   // K = k0+16+8g.. +7
  return __builtin_shufflevector(lo, hi, 0,1,2,3,4,5,6,7,8,9,10,11,12,13,14,15);
}

__device__ __forceinline__ v8f wmma_f16(v16h a, v16h b, v8f c) {
  return __builtin_amdgcn_wmma_f32_16x16x32_f16(false, a, false, b,
                                                (short)0, c, false, false);
}

// ---- software-pipelined 16(M) x 64(N) wave tile ---------------------------
struct Frags { v16h a; v16h b[4]; };

__device__ __forceinline__ void load_frags(Frags& f,
    const _Float16* __restrict__ A, int lda, int m0,
    const _Float16* __restrict__ Bt, int ldk, int n0, int k0) {
  f.a = load_a_frag(A, lda, m0, k0);
  #pragma unroll
  for (int i = 0; i < 4; ++i) f.b[i] = load_b_frag(Bt, ldk, n0 + 16 * i, k0);
}

__device__ __forceinline__ void mma4(v8f* acc, const Frags& f) {
  #pragma unroll
  for (int i = 0; i < 4; ++i) acc[i] = wmma_f16(f.a, f.b[i], acc[i]);
}

// Requires K % 64 == 0. Two-deep pipeline: loads for chunk k+32 are in
// flight while the 4 WMMAs for chunk k execute (partial s_wait_loadcnt,
// back-to-back wmma issue).
__device__ __forceinline__ void wave_tile_gemm(v8f acc[4],
    const _Float16* __restrict__ A, int lda, int m0,
    const _Float16* __restrict__ Bt, int ldk, int n0, int K) {
  Frags f0, f1;
  load_frags(f0, A, lda, m0, Bt, ldk, n0, 0);
  for (int k0 = 0; k0 + 32 < K; k0 += 64) {
    load_frags(f1, A, lda, m0, Bt, ldk, n0, k0 + 32);
    if (k0 + 96 < K) {
      __builtin_prefetch(A  + (size_t)m0 * lda + k0 + 96, 0, 3);
      __builtin_prefetch(Bt + (size_t)n0 * ldk + k0 + 96, 0, 3);
    }
    mma4(acc, f0);
    if (k0 + 64 < K) load_frags(f0, A, lda, m0, Bt, ldk, n0, k0 + 64);
    mma4(acc, f1);
  }
}

// ---- elementwise f32 -> f16 ------------------------------------------------
__global__ __launch_bounds__(256)
void cvt_f32_f16(const float* __restrict__ in, _Float16* __restrict__ out,
                 size_t n) {
  size_t i = (size_t)blockIdx.x * blockDim.x + threadIdx.x;
  size_t stride = (size_t)gridDim.x * blockDim.x;
  for (; i < n; i += stride) out[i] = (_Float16)in[i];
}

// ---- tiled transpose + cvt: W[K x N] f32 -> Wt[N x K] f16 ------------------
__global__ __launch_bounds__(256)
void transpose_cvt_f16(const float* __restrict__ in, _Float16* __restrict__ out,
                       int K, int N) {
  __shared__ float tile[32][33];
  int bx = blockIdx.x * 32;             // K base
  int by = blockIdx.y * 32;             // N base
  int tx = threadIdx.x & 31;
  int ty = threadIdx.x >> 5;            // 8 waves -> 8 row-groups
  #pragma unroll
  for (int i = ty; i < 32; i += 8)
    tile[i][tx] = in[(size_t)(bx + i) * N + by + tx];
  __syncthreads();
  #pragma unroll
  for (int i = ty; i < 32; i += 8)
    out[(size_t)(by + i) * K + bx + tx] = (_Float16)tile[tx][i];
}

// ---- generic WMMA GEMM: C[m,n] = sum_k A16[m,k]*Wt16[n,k] + bias[n] --------
// 256 threads = 8 waves as 4(M) x 2(N); block tile 64(M) x 128(N);
// wave tile 16(M) x 64(N) = 4 accumulators.
__global__ __launch_bounds__(256)
void gemm_f16_bias(const _Float16* __restrict__ A,  int lda,
                   const _Float16* __restrict__ Bt, int ldk,
                   const float* __restrict__ bias,
                   float* __restrict__ C, int ldc, int K) {
  int wave = threadIdx.x >> 5;
  int lane = threadIdx.x & 31;
  int wm = wave & 3, wn = wave >> 2;
  int m0 = blockIdx.x * 64  + wm * 16;
  int n0 = blockIdx.y * 128 + wn * 64;

  v8f acc[4] = {};
  wave_tile_gemm(acc, A, lda, m0, Bt, ldk, n0, K);

  int g = lane >> 4, col = lane & 15;
  #pragma unroll
  for (int i = 0; i < 4; ++i) {
    int n = n0 + 16 * i + col;
    float bv = bias[n];
    #pragma unroll
    for (int r = 0; r < 8; ++r) {
      int m = m0 + r + 8 * g;
      C[(size_t)m * ldc + n] = acc[i][r] + bv;
    }
  }
}

// ---- persistent scan kernel ------------------------------------------------
// grid = 8 blocks x 256 threads. Block b owns N-slice [b*128, b*128+128).
// Each step: hnew[64 x 1024] = leaky_relu(xW_t + hprev @ W_hh), then a
// device-scope arrive-and-spin barrier so every block sees the new h.
__global__ __launch_bounds__(256)
void rnn_scan(const float* __restrict__ xW,        // row = b*T + t, f32
              const _Float16* __restrict__ Whh_t,  // [H x H] pre-transposed
              _Float16* h16a, _Float16* h16b,      // ping-pong h, [B x H]
              float* __restrict__ hiddens,         // d_out part 1, row=b*T+t
              _Float16* __restrict__ hiddens16,    // ws, row = b*T+t
              unsigned* counter, int T, int H) {
  int wave = threadIdx.x >> 5;
  int lane = threadIdx.x & 31;
  int wm = wave & 3, wn = wave >> 2;
  int m0 = wm * 16;                              // batch rows 0..63
  int n0 = blockIdx.x * 128 + wn * 64;
  int g = lane >> 4, col = lane & 15;

  for (int t = 0; t < T; ++t) {
    const _Float16* hprev = (t & 1) ? h16b : h16a;
    _Float16*       hnext = (t & 1) ? h16a : h16b;

    v8f acc[4] = {};
    wave_tile_gemm(acc, hprev, H, m0, Whh_t, H, n0, H);

    #pragma unroll
    for (int i = 0; i < 4; ++i) {
      int n = n0 + 16 * i + col;
      #pragma unroll
      for (int r = 0; r < 8; ++r) {
        int bb = m0 + r + 8 * g;                 // batch index
        size_t row = (size_t)bb * T + t;
        float v = acc[i][r] + xW[row * H + n];
        v = (v > 0.0f) ? v : 0.1f * v;           // leaky_relu(0.1)
        hiddens[row * H + n] = v;
        _Float16 hv = (_Float16)v;
        hnext[(size_t)bb * H + n] = hv;
        hiddens16[row * H + n]    = hv;
      }
    }

    // ---- device-wide step barrier (release writes, arrive, spin, acquire)
    __threadfence();
    __syncthreads();
    if (threadIdx.x == 0) {
      atomicAdd(counter, 1u);
      unsigned target = (unsigned)(t + 1) * gridDim.x;
      while (atomicAdd(counter, 0u) < target) {
        __builtin_amdgcn_s_sleep(2);
      }
    }
    __syncthreads();
    __threadfence();
  }
}

// ---------------------------------------------------------------------------
extern "C" void kernel_launch(void* const* d_in, const int* in_sizes, int n_in,
                              void* d_out, int out_size, void* d_ws,
                              size_t ws_size, hipStream_t stream) {
  (void)in_sizes; (void)n_in; (void)out_size; (void)ws_size;

  const float* x    = (const float*)d_in[0];
  const float* h0   = (const float*)d_in[1];
  /* d_in[2] = c : unused by the reference recurrence */
  const float* W_xh = (const float*)d_in[3];
  const float* W_hh = (const float*)d_in[4];
  const float* b_h  = (const float*)d_in[5];
  const float* W_ho = (const float*)d_in[6];
  const float* b_o  = (const float*)d_in[7];

  const int B = RNN_B, T = RNN_T, D = RNN_D;
  const size_t BT = (size_t)B * T;             // 32768 rows

  float* hiddens = (float*)d_out;              // [B,T,H] row = b*T+t
  float* outputs = hiddens + BT * D;           // [B,T,O]
  float* xW      = outputs;                    // stash xW in outputs region;
                                               // phase-4 GEMM overwrites it.

  // workspace carve-up (all 256B-aligned)
  char* ws = (char*)d_ws;
  _Float16* x16   = (_Float16*)ws;             // 64 MB; reused as hiddens16
  _Float16* Wxh_t = (_Float16*)(ws + BT * D * sizeof(_Float16));
  _Float16* Whh_t = Wxh_t + (size_t)D * D;
  _Float16* Who_t = Whh_t + (size_t)D * D;
  _Float16* h16a  = Who_t + (size_t)D * D;
  _Float16* h16b  = h16a + (size_t)B * D;
  unsigned* counter = (unsigned*)(h16b + (size_t)B * D);
  _Float16* hiddens16 = x16;                   // x16 dead after phase 2

  // phase 1: precision conversion + weight transposes
  cvt_f32_f16<<<2048, 256, 0, stream>>>(x, x16, BT * D);
  dim3 tgrid(D / 32, D / 32);
  transpose_cvt_f16<<<tgrid, 256, 0, stream>>>(W_xh, Wxh_t, D, D);
  transpose_cvt_f16<<<tgrid, 256, 0, stream>>>(W_hh, Whh_t, D, D);
  transpose_cvt_f16<<<tgrid, 256, 0, stream>>>(W_ho, Who_t, D, D);
  cvt_f32_f16<<<64, 256, 0, stream>>>(h0, h16a, (size_t)B * D);
  hipMemsetAsync(counter, 0, sizeof(unsigned), stream);

  // phase 2: xW = x16 @ W_xh + b_h   (M=32768, N=1024, K=1024)
  dim3 ggrid(BT / 64, D / 128);
  gemm_f16_bias<<<ggrid, 256, 0, stream>>>(x16, D, Wxh_t, D, b_h, xW, D, D);

  // phase 3: persistent sequential scan (8 blocks cover N=1024)
  rnn_scan<<<D / 128, 256, 0, stream>>>(xW, Whh_t, h16a, h16b,
                                        hiddens, hiddens16, counter, T, D);

  // phase 4: outputs = hiddens16 @ W_ho + b_o (overwrites xW stash)
  gemm_f16_bias<<<ggrid, 256, 0, stream>>>(hiddens16, D, Who_t, D, b_o,
                                           outputs, D, D);
}
